// spatialAttentionScaledGCN_35699768165242
// MI455X (gfx1250) — matrix operand order repaired and checked by reading, not compiled, not run on hardware
//
#include <hip/hip_runtime.h>
#include <hip/hip_bf16.h>
#include <math.h>

typedef __attribute__((ext_vector_type(16))) _Float16 v16h;
typedef __attribute__((ext_vector_type(8)))  _Float16 v8h;
typedef __attribute__((ext_vector_type(4)))  _Float16 v4h;
typedef __attribute__((ext_vector_type(8)))  float    v8f;

#define B_  32
#define N_  512
#define T_  12
#define F_  64

// Padded LDS leading dimensions: stride % 64 banks == 4 -> conflict-free
// 16-lane tile accesses (each lane starts 4 banks after the previous one).
#define LDK 72    // Ks rows   (64 + 8 f16)
#define LDT 520   // KT rows   (512 + 8 f16)
#define LDW 72    // Ws rows
#define LDP 264   // Ps rows   (256 + 8 f16)
#define LDA 72    // theta-stage A rows (f16)
#define LDO 68    // output-stage rows (f32)

// Byte offsets in dynamic LDS
#define SM_KS 0                               // 512*72*2  = 73728
#define SM_KT 73728                           // 64*520*2  = 66560
#define SM_WS 140288                          // 64*72*2   = 9216
#define SM_PS 149504                          // 8*16*264*2= 67584
#define SM_TOTAL 217088                       // 212 KB of 320 KB

// ---- WMMA operand loaders (wave32, v_wmma_f32_16x16x32_f16 layouts) --------
static __device__ __forceinline__ v16h lds_load_A(const _Float16* base, int ld, int lane) {
  const int m  = lane & 15;
  const int kb = (lane & 16) ? 16 : 0;
  const _Float16* p = base + m * ld + kb;
  v8h lo = *(const v8h*)(p);        // K = kb .. kb+7     (VGPR0-3)
  v8h hi = *(const v8h*)(p + 8);    // K = kb+8 .. kb+15  (VGPR4-7)
  return __builtin_shufflevector(lo, hi, 0,1,2,3,4,5,6,7,8,9,10,11,12,13,14,15);
}

static __device__ __forceinline__ v16h lds_load_B(const _Float16* base, int ld, int lane) {
  const int n  = lane & 15;
  const int ko = (lane & 16) ? 16 : 0;
  return *(const v16h*)(base + n * ld + ko);   // 16 contiguous K values
}

static __device__ __forceinline__ v8f wmma16(v16h a, v16h b, v8f c) {
  return __builtin_amdgcn_wmma_f32_16x16x32_f16(false, a, false, b, (short)0, c, false, false);
}

__global__ __launch_bounds__(256) void sagcn_flash_wmma(
    const float* __restrict__ x, const float* __restrict__ adj,
    const float* __restrict__ Wt, float* __restrict__ out)
{
  extern __shared__ char smem[];
  _Float16* Ks = (_Float16*)(smem + SM_KS);    // [512][LDK]  X rows, f16
  _Float16* KT = (_Float16*)(smem + SM_KT);    // [64][LDT]   X^T, f16
  _Float16* Ws = (_Float16*)(smem + SM_WS);    // [64][LDW]   W_theta, f16
  _Float16* PsAll = (_Float16*)(smem + SM_PS); // [8][16][LDP] per-wave staging

  const int tid  = threadIdx.x;
  const int lane = tid & 31;
  const int wave = tid >> 5;
  const int b    = blockIdx.z;
  const int t    = blockIdx.y;
  const int n0   = blockIdx.x * 128 + wave * 16;

  const size_t xbase = (size_t)b * (N_ * T_ * F_) + (size_t)t * F_;

  // ---------------- Phase 0: stage X (f32->f16, two layouts) + W ------------
  for (int i = tid; i < N_ * (F_ / 4); i += 256) {
    const int n = i >> 4;
    const int f = (i & 15) << 2;
    const float4 v = *(const float4*)(x + xbase + (size_t)n * (T_ * F_) + f);
    const _Float16 h0 = (_Float16)v.x, h1 = (_Float16)v.y,
                   h2 = (_Float16)v.z, h3 = (_Float16)v.w;
    v4h pk = {h0, h1, h2, h3};
    *(v4h*)(Ks + n * LDK + f) = pk;
    KT[(f + 0) * LDT + n] = h0;
    KT[(f + 1) * LDT + n] = h1;
    KT[(f + 2) * LDT + n] = h2;
    KT[(f + 3) * LDT + n] = h3;
  }
  for (int i = tid; i < F_ * (F_ / 4); i += 256) {
    const int o = i >> 4;
    const int f = (i & 15) << 2;
    const float4 v = *(const float4*)(Wt + o * F_ + f);
    v4h pk = {(_Float16)v.x, (_Float16)v.y, (_Float16)v.z, (_Float16)v.w};
    *(v4h*)(Ws + o * LDW + f) = pk;
  }
  __syncthreads();

  _Float16* Ps = PsAll + wave * (16 * LDP);
  const float sc   = 0.125f;                    // 1/sqrt(64)
  const float scl2 = 0.18033688011112042f;      // sc * log2(e)
  const int   col  = lane & 15;
  const int   hi   = lane >> 4;                 // C-layout: VGPR r = row r+8*hi

  float Mrun[8], Srun[8];
#pragma unroll
  for (int r = 0; r < 8; ++r) { Mrun[r] = -3.0e38f; Srun[r] = 0.0f; }
  v8f oac[4];
#pragma unroll
  for (int ft = 0; ft < 4; ++ft) { v8f z = {0,0,0,0,0,0,0,0}; oac[ft] = z; }

  // -------- Flash loop: two 256-column halves with online softmax ----------
#pragma unroll
  for (int h = 0; h < 2; ++h) {
    // ---- S = Q . K^T, B tiles prefetched one tile ahead ----
    v8f s[16];
#pragma unroll
    for (int j = 0; j < 16; ++j) { v8f z = {0,0,0,0,0,0,0,0}; s[j] = z; }
#pragma unroll
    for (int kc = 0; kc < 2; ++kc) {
      const v16h aQ = lds_load_A(Ks + n0 * LDK + kc * 32, LDK, lane);
      v16h bCur = lds_load_B(Ks + (h * 256) * LDK + kc * 32, LDK, lane);
#pragma unroll
      for (int j = 0; j < 16; ++j) {
        v16h bNxt = bCur;
        if (j < 15)
          bNxt = lds_load_B(Ks + (h * 256 + (j + 1) * 16) * LDK + kc * 32, LDK, lane);
        s[j] = wmma16(aQ, bCur, s[j]);
        bCur = bNxt;
      }
    }

    // ---- online softmax: per-row max / sum, rescale running output ----
#pragma unroll
    for (int r = 0; r < 8; ++r) {
      float mx = Mrun[r];
#pragma unroll
      for (int j = 0; j < 16; ++j) mx = fmaxf(mx, s[j][r]);
#pragma unroll
      for (int off = 1; off < 16; off <<= 1)
        mx = fmaxf(mx, __shfl_xor(mx, off, 32));
      const float corr = exp2f(scl2 * (Mrun[r] - mx));
      Mrun[r] = mx;
      float psum = 0.0f;
#pragma unroll
      for (int j = 0; j < 16; ++j) {
        const float e = exp2f(scl2 * (s[j][r] - mx));
        s[j][r] = e;
        psum += e;
      }
#pragma unroll
      for (int off = 1; off < 16; off <<= 1)
        psum += __shfl_xor(psum, off, 32);
      Srun[r] = Srun[r] * corr + psum;
#pragma unroll
      for (int ft = 0; ft < 4; ++ft) oac[ft][r] *= corr;
    }

    // ---- apply adjacency, stage U = adj*e as f16 A-operand in LDS ----
#pragma unroll
    for (int j = 0; j < 16; ++j) {
#pragma unroll
      for (int r = 0; r < 8; ++r) {
        const int row = r + 8 * hi;
        const int m   = h * 256 + j * 16 + col;
        const float a = adj[(size_t)(n0 + row) * N_ + m];   // hot in L2
        Ps[row * LDP + j * 16 + col] = (_Float16)(a * s[j][r]);
      }
    }

    // ---- oac += U . V over this half (A/B prefetched one step ahead) ----
    {
      v16h aP   = lds_load_A(Ps, LDP, lane);
      v16h bCur = lds_load_B(KT + h * 256, LDT, lane);
#pragma unroll
      for (int it = 0; it < 32; ++it) {
        const int kc = it >> 2, ft = it & 3;
        v16h aNxt = aP, bNxt = bCur;
        if (it < 31) {
          const int itn = it + 1;
          const int kcn = itn >> 2, ftn = itn & 3;
          if (ftn == 0)
            aNxt = lds_load_A(Ps + kcn * 32, LDP, lane);
          bNxt = lds_load_B(KT + (ftn * 16) * LDT + h * 256 + kcn * 32, LDT, lane);
        }
        oac[ft] = wmma16(aP, bCur, oac[ft]);
        aP = aNxt; bCur = bNxt;
      }
    }
  }

  // ---- finalize: att = softmax*sc  ->  multiply by sc/Srun ----
#pragma unroll
  for (int r = 0; r < 8; ++r) {
    const float fno = sc / Srun[r];
#pragma unroll
    for (int ft = 0; ft < 4; ++ft) oac[ft][r] *= fno;
  }

  // ---- Theta: stage agg f16 (16 x F_) in Ps, then out = relu(agg . W^T) ----
#pragma unroll
  for (int ft = 0; ft < 4; ++ft) {
#pragma unroll
    for (int r = 0; r < 8; ++r) {
      const int row = r + 8 * hi;
      Ps[row * LDA + ft * 16 + col] = (_Float16)oac[ft][r];
    }
  }
  v8f th[4];
#pragma unroll
  for (int ot = 0; ot < 4; ++ot) { v8f z = {0,0,0,0,0,0,0,0}; th[ot] = z; }
#pragma unroll
  for (int kc = 0; kc < 2; ++kc) {
    const v16h aG = lds_load_A(Ps + kc * 32, LDA, lane);
#pragma unroll
    for (int ot = 0; ot < 4; ++ot) {
      const v16h bW = lds_load_B(Ws + (ot * 16) * LDW + kc * 32, LDW, lane);
      th[ot] = wmma16(aG, bW, th[ot]);
    }
  }

  // ---- ReLU, stage f32 tile in LDS, then coalesced 256B-per-row stores ----
  {
    float* Po = (float*)Ps;                     // 16 x LDO f32 (fits in strip)
#pragma unroll
    for (int ot = 0; ot < 4; ++ot) {
#pragma unroll
      for (int r = 0; r < 8; ++r) {
        const int row = r + 8 * hi;
        Po[row * LDO + ot * 16 + col] = fmaxf(th[ot][r], 0.0f);
      }
    }
#pragma unroll
    for (int i = lane; i < 256; i += 32) {      // 16 rows x 16 float4
      const int row = i >> 4;
      const int f4  = (i & 15) << 2;
      const float4 v = *(const float4*)(Po + row * LDO + f4);
      *(float4*)(out + xbase + (size_t)(n0 + row) * (T_ * F_) + f4) = v;
    }
  }
}

extern "C" void kernel_launch(void* const* d_in, const int* in_sizes, int n_in,
                              void* d_out, int out_size, void* d_ws, size_t ws_size,
                              hipStream_t stream) {
  (void)in_sizes; (void)n_in; (void)out_size; (void)d_ws; (void)ws_size;
  const float* x   = (const float*)d_in[0];
  const float* adj = (const float*)d_in[1];
  const float* Wt  = (const float*)d_in[2];
  float* out = (float*)d_out;
  dim3 grid(N_ / 128, T_, B_);   // (4, 12, 32) = 1536 workgroups
  sagcn_flash_wmma<<<grid, 256, SM_TOTAL, stream>>>(x, adj, Wt, out);
}
